// TransformerBlock_35682588295340
// MI455X (gfx1250) — compile-verified, matrix operand order checked by reading, not run on hardware
//
#include <hip/hip_runtime.h>
#include <stdint.h>

#define D_MODEL 1024
#define N_HEADS 16
#define D_HEAD  64
#define D_FF    4096
#define SEQ     2048
#define BATCH   4
#define NROWS   (BATCH * SEQ)   // 8192

typedef __bf16 bf16_t;
typedef __attribute__((ext_vector_type(8)))  __bf16 v8bf;
typedef __attribute__((ext_vector_type(16))) __bf16 v16bf;
typedef __attribute__((ext_vector_type(8)))  float  v8f;

__device__ __forceinline__ unsigned short f32_to_bf16_bits(float f) {
    unsigned u = __float_as_uint(f);
    u = (u + 0x7FFFu + ((u >> 16) & 1u)) >> 16;   // round-to-nearest-even
    return (unsigned short)u;
}
__device__ __forceinline__ bf16_t to_bf16(float f) {
    union { unsigned short u; bf16_t b; } cv;
    cv.u = f32_to_bf16_bits(f);
    return cv.b;
}
__device__ __forceinline__ v16bf make_frag(v8bf lo, v8bf hi) {
    union { v16bf v; v8bf h[2]; } u;
    u.h[0] = lo; u.h[1] = hi;
    return u.v;
}

// 16x32 bf16 fragment in the CDNA5 16-bit A/B VGPR layout, from a row-major
// matrix. p points at (row0, k0). Lane L holds row (L&15); elems[0..7] cover
// k = 8*(L>>4)..+7, elems[8..15] cover k = 16+8*(L>>4)..+7. Two b128 loads.
__device__ __forceinline__ v16bf load_frag(const bf16_t* p, int ld, int lane) {
    const bf16_t* q = p + (size_t)(lane & 15) * ld + ((lane >> 4) << 3);
    v8bf lo = *(const v8bf*)q;
    v8bf hi = *(const v8bf*)(q + 16);
    return make_frag(lo, hi);
}

__device__ __forceinline__ v8f wmma_bf16(v16bf a, v16bf b, v8f c) {
    // (neg_a, A, neg_b, B, c_mod, C, reuse_a, reuse_b)
    return __builtin_amdgcn_wmma_f32_16x16x32_bf16(false, a, false, b, (short)0, c,
                                                   false, false);
}

// ---------------------------------------------------------------- weight pack
__global__ void pack_qkv_w_kernel(const float* __restrict__ wq,
                                  const float* __restrict__ wk,
                                  const float* __restrict__ wv,
                                  unsigned short* __restrict__ dst) {
    // dst[j][d] (j = sel*1024 + h*64 + e) = w_sel[h][d][e]
    size_t id = (size_t)blockIdx.x * blockDim.x + threadIdx.x;
    if (id >= (size_t)3072 * 1024) return;
    int d  = (int)(id & 1023);
    int j  = (int)(id >> 10);
    int sel = j >> 10;
    int he  = j & 1023;
    int h = he >> 6, e = he & 63;
    const float* w = (sel == 0) ? wq : ((sel == 1) ? wk : wv);
    dst[id] = f32_to_bf16_bits(w[((size_t)h * 1024 + d) * 64 + e]);
}

__global__ void transpose_pack_kernel(const float* __restrict__ src,
                                      unsigned short* __restrict__ dst,
                                      int R, int C) {
    // src [R][C] f32 -> dst [C][R] bf16
    size_t id = (size_t)blockIdx.x * blockDim.x + threadIdx.x;   // id = c*R + r
    if (id >= (size_t)R * C) return;
    int r = (int)(id % R);
    int c = (int)(id / R);
    dst[id] = f32_to_bf16_bits(src[(size_t)r * C + c]);
}

__global__ void pack_qkv_bias_kernel(const float* __restrict__ bq,
                                     const float* __restrict__ bk,
                                     const float* __restrict__ bv,
                                     float* __restrict__ dst) {
    int j = blockIdx.x * blockDim.x + threadIdx.x;
    if (j >= 3072) return;
    dst[j] = (j < 1024) ? bq[j] : ((j < 2048) ? bk[j - 1024] : bv[j - 2048]);
}

// ------------------------------------------------------------------ layernorm
__global__ __launch_bounds__(256)
void layernorm_bf16_kernel(const float* __restrict__ x,
                           const float* __restrict__ g,
                           const float* __restrict__ bta,
                           unsigned short* __restrict__ out) {
    const int row = blockIdx.x;
    const float* xr = x + (size_t)row * D_MODEL;
    float s = 0.f, s2 = 0.f;
    for (int i = threadIdx.x; i < D_MODEL; i += 256) {
        float v = xr[i];
        s += v; s2 += v * v;
    }
    for (int m = 16; m >= 1; m >>= 1) {
        s  += __shfl_xor(s,  m, 32);
        s2 += __shfl_xor(s2, m, 32);
    }
    __shared__ float ws1[8], ws2[8];
    int wave = threadIdx.x >> 5;
    if ((threadIdx.x & 31) == 0) { ws1[wave] = s; ws2[wave] = s2; }
    __syncthreads();
    s = 0.f; s2 = 0.f;
    for (int w = 0; w < 8; ++w) { s += ws1[w]; s2 += ws2[w]; }
    const float mu   = s * (1.0f / D_MODEL);
    const float var  = s2 * (1.0f / D_MODEL) - mu * mu;
    const float rstd = rsqrtf(var + 1e-5f);
    for (int i = threadIdx.x; i < D_MODEL; i += 256)
        out[(size_t)row * D_MODEL + i] =
            f32_to_bf16_bits((xr[i] - mu) * rstd * g[i] + bta[i]);
}

// ----------------------------------------------------------------- WMMA GEMM
// C[M,N] = A[M,K](bf16,rowmajor) * BT[N,K]^T (+bias)(+relu)(+resid f32)
// 8 waves as 4(M) x 2(N); wave tile 32x64 = 2x4 WMMA C-tiles; block 128x128.
// Software-pipelined: next K-step's 6 fragments are issued before the 8 WMMAs
// of the current K-step so loads overlap matrix ops.
__global__ __launch_bounds__(256)
void gemm_bf16_kernel(const bf16_t* __restrict__ A, const bf16_t* __restrict__ BT,
                      const float* __restrict__ bias, const float* resid,
                      float* outF, unsigned short* __restrict__ outB,
                      int M, int N, int K, int relu) {
    const int wave = threadIdx.x >> 5;
    const int lane = threadIdx.x & 31;
    const int wm = wave & 3;          // 4 waves along M
    const int wn = wave >> 2;         // 2 waves along N
    const int rowBase = blockIdx.y * 128 + wm * 32;
    const int colBase = blockIdx.x * 128 + wn * 64;

    v8f acc[2][4];
#pragma unroll
    for (int mi = 0; mi < 2; ++mi)
#pragma unroll
        for (int ni = 0; ni < 4; ++ni)
#pragma unroll
            for (int r = 0; r < 8; ++r) acc[mi][ni][r] = 0.f;

    const bf16_t* Ap0 = A  + (size_t)rowBase        * K;
    const bf16_t* Ap1 = A  + (size_t)(rowBase + 16) * K;
    const bf16_t* Bp0 = BT + (size_t)(colBase +  0) * K;
    const bf16_t* Bp1 = BT + (size_t)(colBase + 16) * K;
    const bf16_t* Bp2 = BT + (size_t)(colBase + 32) * K;
    const bf16_t* Bp3 = BT + (size_t)(colBase + 48) * K;

    // prologue: k0 = 0
    v16bf a0 = load_frag(Ap0, K, lane), a1 = load_frag(Ap1, K, lane);
    v16bf b0 = load_frag(Bp0, K, lane), b1 = load_frag(Bp1, K, lane);
    v16bf b2 = load_frag(Bp2, K, lane), b3 = load_frag(Bp3, K, lane);

    for (int k0 = 32; k0 < K; k0 += 32) {
        if (k0 + 96 < K) {   // prefetch ahead -> global_prefetch_b8
            __builtin_prefetch(Ap0 + (size_t)(lane & 15) * K + k0 + 96, 0, 3);
            __builtin_prefetch(Bp0 + (size_t)(lane & 15) * K + k0 + 96, 0, 3);
        }
        // issue next K-step loads first (overlap with WMMA below)
        v16bf na0 = load_frag(Ap0 + k0, K, lane);
        v16bf na1 = load_frag(Ap1 + k0, K, lane);
        v16bf nb0 = load_frag(Bp0 + k0, K, lane);
        v16bf nb1 = load_frag(Bp1 + k0, K, lane);
        v16bf nb2 = load_frag(Bp2 + k0, K, lane);
        v16bf nb3 = load_frag(Bp3 + k0, K, lane);
        acc[0][0] = wmma_bf16(a0, b0, acc[0][0]);
        acc[0][1] = wmma_bf16(a0, b1, acc[0][1]);
        acc[0][2] = wmma_bf16(a0, b2, acc[0][2]);
        acc[0][3] = wmma_bf16(a0, b3, acc[0][3]);
        acc[1][0] = wmma_bf16(a1, b0, acc[1][0]);
        acc[1][1] = wmma_bf16(a1, b1, acc[1][1]);
        acc[1][2] = wmma_bf16(a1, b2, acc[1][2]);
        acc[1][3] = wmma_bf16(a1, b3, acc[1][3]);
        a0 = na0; a1 = na1;
        b0 = nb0; b1 = nb1; b2 = nb2; b3 = nb3;
    }
    // epilogue K-step
    acc[0][0] = wmma_bf16(a0, b0, acc[0][0]);
    acc[0][1] = wmma_bf16(a0, b1, acc[0][1]);
    acc[0][2] = wmma_bf16(a0, b2, acc[0][2]);
    acc[0][3] = wmma_bf16(a0, b3, acc[0][3]);
    acc[1][0] = wmma_bf16(a1, b0, acc[1][0]);
    acc[1][1] = wmma_bf16(a1, b1, acc[1][1]);
    acc[1][2] = wmma_bf16(a1, b2, acc[1][2]);
    acc[1][3] = wmma_bf16(a1, b3, acc[1][3]);

    const int hrow = (lane >> 4) << 3;   // C layout: row = hrow + r, col = lane&15
    const int ncol = lane & 15;
#pragma unroll
    for (int mi = 0; mi < 2; ++mi) {
#pragma unroll
        for (int ni = 0; ni < 4; ++ni) {
            int col = colBase + ni * 16 + ncol;
            float bv = bias ? bias[col] : 0.0f;
#pragma unroll
            for (int r = 0; r < 8; ++r) {
                int row = rowBase + mi * 16 + hrow + r;
                float v = acc[mi][ni][r] + bv;
                if (relu)  v = fmaxf(v, 0.0f);
                if (resid) v += resid[(size_t)row * N + col];
                if (outF)  outF[(size_t)row * N + col] = v;
                else       outB[(size_t)row * N + col] = f32_to_bf16_bits(v);
            }
        }
    }
}

// ---------------------------------------------------- flash causal attention
// grid (SEQ/64, H, B), 128 threads = 4 waves; wave w owns 16 query rows.
__global__ __launch_bounds__(128)
void attn_kernel(const bf16_t* __restrict__ qkv, unsigned short* __restrict__ o_out) {
    __shared__ bf16_t sVT[64 * 72];      // V^T chunk [e][t], padded stride 72
    __shared__ bf16_t sP[4 * 16 * 72];   // per-wave P tile [m][t], padded

    const int wave  = threadIdx.x >> 5;
    const int lane  = threadIdx.x & 31;
    const int qtile = blockIdx.x;
    const int h     = blockIdx.y;
    const int b     = blockIdx.z;
    const int qbase = qtile * 64 + wave * 16;
    const size_t rs = 3 * D_MODEL;       // qkv row stride (q|k|v concat)

    v16bf qf0 = load_frag(qkv + (size_t)(b * SEQ + qbase) * rs + h * 64 + 0,  (int)rs, lane);
    v16bf qf1 = load_frag(qkv + (size_t)(b * SEQ + qbase) * rs + h * 64 + 32, (int)rs, lane);

    v8f oacc[4];
#pragma unroll
    for (int ni = 0; ni < 4; ++ni)
#pragma unroll
        for (int r = 0; r < 8; ++r) oacc[ni][r] = 0.f;
    float mrun[8], lrun[8];
#pragma unroll
    for (int r = 0; r < 8; ++r) { mrun[r] = -1e30f; lrun[r] = 0.f; }

    const int hrow = (lane >> 4) << 3;
    const int ncol = lane & 15;
    const int nchunks = qtile + 1;

    for (int c = 0; c < nchunks; ++c) {
        const int tbase = c * 64;
        __syncthreads();   // previous iteration's LDS reads complete
        // stage V chunk transposed into LDS (coalesced reads, VT[e][t] writes)
        for (int i = threadIdx.x; i < 64 * 64; i += 128) {
            int t = i >> 6, e = i & 63;
            sVT[e * 72 + t] =
                qkv[(size_t)(b * SEQ + tbase + t) * rs + 2 * D_MODEL + h * 64 + e];
        }
        // scores = Q K^T * 1/sqrt(dh), causal mask
        v8f sc[4];
#pragma unroll
        for (int nt = 0; nt < 4; ++nt) {
            v8f s;
#pragma unroll
            for (int r = 0; r < 8; ++r) s[r] = 0.f;
            const bf16_t* kb =
                qkv + (size_t)(b * SEQ + tbase + nt * 16) * rs + D_MODEL + h * 64;
            v16bf kf0 = load_frag(kb,      (int)rs, lane);
            v16bf kf1 = load_frag(kb + 32, (int)rs, lane);
            s = wmma_bf16(qf0, kf0, s);
            s = wmma_bf16(qf1, kf1, s);
#pragma unroll
            for (int r = 0; r < 8; ++r) {
                int row = qbase + hrow + r;
                int col = tbase + nt * 16 + ncol;
                sc[nt][r] = (col <= row) ? s[r] * 0.125f : -1e30f;
            }
        }
        // online softmax (row reductions across the 16-lane half groups)
#pragma unroll
        for (int r = 0; r < 8; ++r) {
            float cm = fmaxf(fmaxf(sc[0][r], sc[1][r]), fmaxf(sc[2][r], sc[3][r]));
            for (int m = 1; m < 16; m <<= 1) cm = fmaxf(cm, __shfl_xor(cm, m, 32));
            float nm   = fmaxf(mrun[r], cm);
            float corr = __expf(mrun[r] - nm);
            mrun[r] = nm;
            float ps = 0.f;
#pragma unroll
            for (int nt = 0; nt < 4; ++nt) {
                float p = __expf(sc[nt][r] - nm);
                sc[nt][r] = p; ps += p;
            }
            for (int m = 1; m < 16; m <<= 1) ps += __shfl_xor(ps, m, 32);
            lrun[r] = lrun[r] * corr + ps;
#pragma unroll
            for (int ni = 0; ni < 4; ++ni) oacc[ni][r] *= corr;
        }
        // redistribute P: C-layout -> LDS -> A-layout fragments
#pragma unroll
        for (int nt = 0; nt < 4; ++nt)
#pragma unroll
            for (int r = 0; r < 8; ++r)
                sP[wave * 1152 + (hrow + r) * 72 + nt * 16 + ncol] = to_bf16(sc[nt][r]);
        __syncthreads();   // sVT + sP visible
        // O += P x V
#pragma unroll
        for (int kk = 0; kk < 2; ++kk) {
            v16bf pf = load_frag(sP + wave * 1152 + kk * 32, 72, lane);
#pragma unroll
            for (int ni = 0; ni < 4; ++ni) {
                v16bf vf = load_frag(sVT + (ni * 16) * 72 + kk * 32, 72, lane);
                oacc[ni] = wmma_bf16(pf, vf, oacc[ni]);
            }
        }
    }

    // normalize, write o[b, s, h*64 + e] (concat-heads layout)
#pragma unroll
    for (int ni = 0; ni < 4; ++ni) {
#pragma unroll
        for (int r = 0; r < 8; ++r) {
            int row = qbase + hrow + r;
            int e   = ni * 16 + ncol;
            float v = oacc[ni][r] / lrun[r];
            o_out[(size_t)(b * SEQ + row) * D_MODEL + h * 64 + e] = f32_to_bf16_bits(v);
        }
    }
}

// -------------------------------------------------------------------- driver
extern "C" void kernel_launch(void* const* d_in, const int* in_sizes, int n_in,
                              void* d_out, int out_size, void* d_ws, size_t ws_size,
                              hipStream_t stream) {
    (void)in_sizes; (void)n_in; (void)out_size; (void)ws_size;
    const float* x     = (const float*)d_in[0];
    const float* ln1_g = (const float*)d_in[1];
    const float* ln1_b = (const float*)d_in[2];
    const float* wq    = (const float*)d_in[3];
    const float* bq    = (const float*)d_in[4];
    const float* wk    = (const float*)d_in[5];
    const float* bk    = (const float*)d_in[6];
    const float* wv    = (const float*)d_in[7];
    const float* bv    = (const float*)d_in[8];
    const float* wp    = (const float*)d_in[9];
    const float* bp    = (const float*)d_in[10];
    const float* ln2_g = (const float*)d_in[11];
    const float* ln2_b = (const float*)d_in[12];
    const float* w1    = (const float*)d_in[13];
    const float* b1    = (const float*)d_in[14];
    const float* w2    = (const float*)d_in[15];
    const float* b2    = (const float*)d_in[16];
    float* out = (float*)d_out;

    // workspace layout (bytes); total = 109,117,440
    char* ws = (char*)d_ws;
    unsigned short* wqkvT = (unsigned short*)(ws);              // [3072,1024] bf16
    unsigned short* wpT   = (unsigned short*)(ws +  6291456);   // [1024,1024]
    unsigned short* w1T   = (unsigned short*)(ws +  8388608);   // [4096,1024]
    unsigned short* w2T   = (unsigned short*)(ws + 16777216);   // [1024,4096]
    float*          biasQ = (float*)         (ws + 25165824);   // [3072] f32
    unsigned short* xn    = (unsigned short*)(ws + 25231360);   // [8192,1024]
    unsigned short* qkv   = (unsigned short*)(ws + 42008576);   // [8192,3072]
    unsigned short* oatt  = (unsigned short*)(ws + 92340224);   // [8192,1024]
    unsigned short* hbuf  = (unsigned short*)(ws + 42008576);   // [8192,4096] (reuses qkv+oatt)

    dim3 b256(256);
    pack_qkv_w_kernel<<<(3072 * 1024 + 255) / 256, b256, 0, stream>>>(wq, wk, wv, wqkvT);
    transpose_pack_kernel<<<(1024 * 1024 + 255) / 256, b256, 0, stream>>>(wp, wpT, 1024, 1024);
    transpose_pack_kernel<<<(1024 * 4096 + 255) / 256, b256, 0, stream>>>(w1, w1T, 1024, 4096);
    transpose_pack_kernel<<<(4096 * 1024 + 255) / 256, b256, 0, stream>>>(w2, w2T, 4096, 1024);
    pack_qkv_bias_kernel<<<12, b256, 0, stream>>>(bq, bk, bv, biasQ);

    // xn = LN1(x) in bf16
    layernorm_bf16_kernel<<<NROWS, b256, 0, stream>>>(x, ln1_g, ln1_b, xn);

    // fused QKV projection: [8192,1024] x [1024,3072]
    gemm_bf16_kernel<<<dim3(3072 / 128, NROWS / 128), b256, 0, stream>>>(
        (const bf16_t*)xn, (const bf16_t*)wqkvT, biasQ, nullptr, nullptr, qkv,
        NROWS, 3072, 1024, 0);

    // causal flash attention -> oatt (concat heads)
    attn_kernel<<<dim3(SEQ / 64, N_HEADS, BATCH), dim3(128), 0, stream>>>(
        (const bf16_t*)qkv, oatt);

    // out = x + oatt @ wp + bp
    gemm_bf16_kernel<<<dim3(1024 / 128, NROWS / 128), b256, 0, stream>>>(
        (const bf16_t*)oatt, (const bf16_t*)wpT, bp, x, out, nullptr,
        NROWS, 1024, 1024, 0);

    // xn = LN2(out)
    layernorm_bf16_kernel<<<NROWS, b256, 0, stream>>>(out, ln2_g, ln2_b, xn);

    // h = relu(xn @ w1 + b1)
    gemm_bf16_kernel<<<dim3(4096 / 128, NROWS / 128), b256, 0, stream>>>(
        (const bf16_t*)xn, (const bf16_t*)w1T, b1, nullptr, nullptr, hbuf,
        NROWS, 4096, 1024, 1);

    // out = out + h @ w2 + b2
    gemm_bf16_kernel<<<dim3(1024 / 128, NROWS / 128), b256, 0, stream>>>(
        (const bf16_t*)hbuf, (const bf16_t*)w2T, b2, out, out, nullptr,
        NROWS, 1024, 4096, 0);
}
